// GNN_12369505813069
// MI455X (gfx1250) — matrix operand order repaired and checked by reading, not compiled, not run on hardware
//
#include <hip/hip_runtime.h>
#include <math.h>

typedef long long i64;
typedef __bf16 bf16;
typedef __attribute__((ext_vector_type(16))) __bf16 v16bf;
typedef __attribute__((ext_vector_type(8)))  float  v8f;

#define WGROWS 64   // rows of X per workgroup in the WMMA GEMM
#define KPAD   8    // bf16 elements of LDS row padding (16B) to break bank conflicts

// ---------------------------------------------------------------- utilities

__global__ void k_zero_f32(float* p, i64 n) {
  i64 i = (i64)blockIdx.x * blockDim.x + threadIdx.x;
  i64 stride = (i64)gridDim.x * blockDim.x;
  for (; i < n; i += stride) p[i] = 0.0f;
}

__global__ void k_convert_bf16(const float* __restrict__ src, bf16* __restrict__ dst, int n) {
  int i = blockIdx.x * blockDim.x + threadIdx.x;
  if (i < n) dst[i] = (bf16)src[i];
}

__global__ void k_init_emb(const float* __restrict__ init, float* __restrict__ emb,
                           bf16* __restrict__ embb, int N, int D, float scale) {
  i64 i = (i64)blockIdx.x * blockDim.x + threadIdx.x;
  if (i < (i64)N * D) {
    float v = init[(int)(i % D)] * scale;
    emb[i] = v;
    embb[i] = (bf16)v;
  }
}

__global__ void k_degree(const int* __restrict__ idx, float* __restrict__ deg, int E) {
  int i = blockIdx.x * blockDim.x + threadIdx.x;
  if (i < E) atomicAdd(&deg[idx[i]], 1.0f);
}

// ---------------------------------------------------------------- WMMA GEMM
// Y[N,M] = act(X[N,K] @ W[M,K]^T + bias), X/W bf16, accumulate f32.
// 4 waves per block; each wave computes a 16-row strip with v_wmma_f32_16x16x32_bf16.
__global__ __launch_bounds__(128)
void k_gemm(const bf16* __restrict__ X, const bf16* __restrict__ W,
            const float* __restrict__ bias,
            float* __restrict__ Yf, bf16* __restrict__ Yb,
            int N, int K, int M, int relu) {
  __shared__ bf16 lds[WGROWS * (256 + KPAD)];
  const int Kp = K + KPAD;
  const int tid = threadIdx.x;
  const int rowBase = blockIdx.x * WGROWS;

  // Cooperative, coalesced load of the X tile (WGROWS x K) into LDS; OOB rows -> 0.
  {
    unsigned* l32 = (unsigned*)lds;
    const unsigned* g32 = (const unsigned*)X;
    const int dwRow = K >> 1;
    const int total = WGROWS * dwRow;
    for (int idx = tid; idx < total; idx += 128) {
      int r = idx / dwRow, c = idx - r * dwRow;
      int gr = rowBase + r;
      unsigned v = 0u;
      if (gr < N) v = g32[(i64)gr * dwRow + c];
      l32[r * (Kp >> 1) + c] = v;
    }
  }
  __syncthreads();

  const int wave  = tid >> 5;
  const int lane  = tid & 31;
  const int lhalf = lane >> 4;   // 0: lanes 0-15, 1: lanes 16-31
  const int lmod  = lane & 15;
  const int r0    = wave * 16;

  const int nTiles = M >> 4;
  const int kTiles = K >> 5;

  for (int nt = 0; nt < nTiles; ++nt) {
    const int col = (nt << 4) + lmod;
    const float bv = bias ? bias[col] : 0.0f;
    v8f acc = {bv, bv, bv, bv, bv, bv, bv, bv};

    for (int kt = 0; kt < kTiles; ++kt) {
      // A fragment: contiguous k order, same as B (lane%16 = M row; lanes 16-31 read K+16; 2 halfs/VGPR)
      union { v16bf v; unsigned u[8]; } A;
      const int lrow = r0 + lmod;
      #pragma unroll
      for (int t = 0; t < 8; ++t) {
        int k = (kt << 5) + (lhalf << 4) + (t << 1);
        A.u[t] = *(const unsigned*)&lds[lrow * Kp + k];
      }
      // B fragment: B[k][n] = W[n][k]; each lane reads 16 contiguous bf16 of W row `col`
      const bf16* wp = W + (i64)col * K + (kt << 5) + (lhalf << 4);
      v16bf Bf = *(const v16bf*)wp;

      acc = __builtin_amdgcn_wmma_f32_16x16x32_bf16(false, A.v, false, Bf,
                                                    (short)0, acc, false, false);
    }

    // Epilogue per C/D layout: col = lane%16, row = VGPR + 8*(lane/16)
    #pragma unroll
    for (int v = 0; v < 8; ++v) {
      int row = rowBase + r0 + v + (lhalf << 3);
      if (row < N) {
        float val = acc[v];
        if (relu) val = fmaxf(val, 0.0f);
        if (Yf) Yf[(i64)row * M + col] = val;
        if (Yb) Yb[(i64)row * M + col] = (bf16)val;
      }
    }
  }
}

// ------------------------------------------------------- scatter / pointwise

__global__ void k_scatter(const float* __restrict__ msg, const int* __restrict__ srcIdx,
                          const int* __restrict__ dstIdx, float* __restrict__ aggr,
                          int E, int D) {
  const int chunks = D >> 2;
  i64 i = (i64)blockIdx.x * blockDim.x + threadIdx.x;
  i64 total = (i64)E * chunks;
  i64 stride = (i64)gridDim.x * blockDim.x;
  for (; i < total; i += stride) {
    int e = (int)(i / chunks);
    int c = (int)(i - (i64)e * chunks);
    const float4 m = ((const float4*)(msg + (i64)srcIdx[e] * D))[c];
    float* a = aggr + (i64)dstIdx[e] * D + (c << 2);
    atomicAdd(a + 0, m.x); atomicAdd(a + 1, m.y);
    atomicAdd(a + 2, m.z); atomicAdd(a + 3, m.w);
  }
}

__global__ void k_div_deg_bf16(const float* __restrict__ aggr, const float* __restrict__ deg,
                               bf16* __restrict__ out, int N, int D) {
  i64 i = (i64)blockIdx.x * blockDim.x + threadIdx.x;
  if (i < (i64)N * D) {
    int r = (int)(i / D);
    out[i] = (bf16)(aggr[i] / fmaxf(deg[r], 1.0f));
  }
}

// xbuf[i] = concat(c2l_aggr[i]/deg, l_emb[i^1])  (negation-pair swap)
__global__ void k_build_lgru_x(const float* __restrict__ aggr, const float* __restrict__ deg,
                               const float* __restrict__ hl, bf16* __restrict__ xbuf,
                               int Lr, int D) {
  i64 i = (i64)blockIdx.x * blockDim.x + threadIdx.x;
  if (i < (i64)Lr * D) {
    int r = (int)(i / D), d = (int)(i - (i64)r * D);
    xbuf[(i64)r * (2 * D) + d]     = (bf16)(aggr[i] / fmaxf(deg[r], 1.0f));
    xbuf[(i64)r * (2 * D) + D + d] = (bf16)hl[(i64)(r ^ 1) * D + d];
  }
}

__device__ __forceinline__ float sigm(float x) { return 1.0f / (1.0f + __expf(-x)); }

__global__ void k_gru(const float* __restrict__ gi, const float* __restrict__ gh,
                      float* __restrict__ h, bf16* __restrict__ hb, int N, int D) {
  i64 i = (i64)blockIdx.x * blockDim.x + threadIdx.x;
  if (i < (i64)N * D) {
    int r = (int)(i / D), d = (int)(i - (i64)r * D);
    i64 b = (i64)r * 3 * D + d;
    float rg = sigm(gi[b] + gh[b]);
    float zg = sigm(gi[b + D] + gh[b + D]);
    float ng = tanhf(gi[b + 2 * D] + rg * gh[b + 2 * D]);
    float nh = (1.0f - zg) * ng + zg * h[i];
    h[i]  = nh;
    hb[i] = (bf16)nh;
  }
}

__global__ void k_readout(const bf16* __restrict__ hid, const float* __restrict__ w2,
                          const float* __restrict__ b2, const int* __restrict__ lbatch,
                          float* __restrict__ gsum, float* __restrict__ gcnt, int Lr, int D) {
  int i = blockIdx.x * blockDim.x + threadIdx.x;
  if (i < Lr) {
    float s = b2[0];
    const bf16* hp = hid + (i64)i * D;
    for (int d = 0; d < D; ++d) s += (float)hp[d] * w2[d];
    int b = lbatch[i];
    atomicAdd(&gsum[b], s);
    atomicAdd(&gcnt[b], 1.0f);
  }
}

__global__ void k_final(const float* __restrict__ gsum, const float* __restrict__ gcnt,
                        float* __restrict__ out, int B) {
  int i = threadIdx.x;
  if (i < B) out[i] = 1.0f / (1.0f + __expf(-gsum[i] / fmaxf(gcnt[i], 1.0f)));
}

// ---------------------------------------------------------------- launcher

extern "C" void kernel_launch(void* const* d_in, const int* in_sizes, int n_in,
                              void* d_out, int out_size, void* d_ws, size_t ws_size,
                              hipStream_t stream) {
  const int E  = in_sizes[0];
  const int Lr = in_sizes[2];
  const int D  = in_sizes[3];     // 128
  const int C  = 100000;          // segment count; fixed in reference, not derivable from inputs
  const int B  = out_size;
  const int NITER = 4;

  const int* lEdge  = (const int*)d_in[0];
  const int* cEdge  = (const int*)d_in[1];
  const int* lBatch = (const int*)d_in[2];
  const float* lInit = (const float*)d_in[3];
  const float* cInit = (const float*)d_in[4];
  const float* l2c_w1 = (const float*)d_in[5];  const float* l2c_b1 = (const float*)d_in[6];
  const float* l2c_w2 = (const float*)d_in[7];  const float* l2c_b2 = (const float*)d_in[8];
  const float* c2l_w1 = (const float*)d_in[9];  const float* c2l_b1 = (const float*)d_in[10];
  const float* c2l_w2 = (const float*)d_in[11]; const float* c2l_b2 = (const float*)d_in[12];
  const float* cg_wih = (const float*)d_in[13]; const float* cg_whh = (const float*)d_in[14];
  const float* cg_bih = (const float*)d_in[15]; const float* cg_bhh = (const float*)d_in[16];
  const float* lg_wih = (const float*)d_in[17]; const float* lg_whh = (const float*)d_in[18];
  const float* lg_bih = (const float*)d_in[19]; const float* lg_bhh = (const float*)d_in[20];
  const float* ro_w1  = (const float*)d_in[21]; const float* ro_b1  = (const float*)d_in[22];
  const float* ro_w2  = (const float*)d_in[23]; const float* ro_b2  = (const float*)d_in[24];

  char* ws = (char*)d_ws;
  size_t off = 0;
  auto carve = [&](size_t bytes) -> void* {
    off = (off + 255) & ~(size_t)255;
    void* p = ws + off; off += bytes; return p;
  };
  const int NMAX = (C > Lr) ? C : Lr;

  float* hl   = (float*)carve((size_t)Lr * D * 4);
  float* hc   = (float*)carve((size_t)C  * D * 4);
  bf16*  hlb  = (bf16*) carve((size_t)Lr * D * 2);
  bf16*  hcb  = (bf16*) carve((size_t)C  * D * 2);
  float* ldeg = (float*)carve((size_t)Lr * 4);
  float* cdeg = (float*)carve((size_t)C  * 4);

  bf16* wb_l2c1 = (bf16*)carve((size_t)D * D * 2);
  bf16* wb_l2c2 = (bf16*)carve((size_t)D * D * 2);
  bf16* wb_c2l1 = (bf16*)carve((size_t)D * D * 2);
  bf16* wb_c2l2 = (bf16*)carve((size_t)D * D * 2);
  bf16* wb_cwih = (bf16*)carve((size_t)3 * D * D * 2);
  bf16* wb_cwhh = (bf16*)carve((size_t)3 * D * D * 2);
  bf16* wb_lwih = (bf16*)carve((size_t)3 * D * 2 * D * 2);
  bf16* wb_lwhh = (bf16*)carve((size_t)3 * D * D * 2);
  bf16* wb_ro1  = (bf16*)carve((size_t)D * D * 2);

  size_t xElems = (size_t)Lr * 2 * D; if ((size_t)C * D > xElems) xElems = (size_t)C * D;
  bf16*  xbuf = (bf16*) carve(xElems * 2);
  bf16*  hbuf = (bf16*) carve((size_t)NMAX * D * 2);
  float* mbuf = (float*)carve((size_t)NMAX * D * 4);
  float* abuf = (float*)carve((size_t)NMAX * D * 4);
  float* g1   = (float*)carve((size_t)NMAX * 3 * D * 4);
  float* g2   = (float*)carve((size_t)NMAX * 3 * D * 4);
  float* gsum = (float*)carve((size_t)B * 4);
  float* gcnt = (float*)carve((size_t)B * 4);

  auto cdiv = [](i64 a, i64 b) -> int { return (int)((a + b - 1) / b); };
  auto zero = [&](float* p, i64 n) {
    k_zero_f32<<<2048, 256, 0, stream>>>(p, n);
  };
  auto gemm = [&](const bf16* X, const bf16* W, const float* bias,
                  float* Yf, bf16* Yb, int N, int K, int M, int relu) {
    k_gemm<<<cdiv(N, WGROWS), 128, 0, stream>>>(X, W, bias, Yf, Yb, N, K, M, relu);
  };

  // one-time (per launch) prep: weight conversion, degrees, init embeddings
  struct { const float* s; bf16* d; int n; } wc[9] = {
    {l2c_w1, wb_l2c1, D * D}, {l2c_w2, wb_l2c2, D * D},
    {c2l_w1, wb_c2l1, D * D}, {c2l_w2, wb_c2l2, D * D},
    {cg_wih, wb_cwih, 3 * D * D}, {cg_whh, wb_cwhh, 3 * D * D},
    {lg_wih, wb_lwih, 3 * D * 2 * D}, {lg_whh, wb_lwhh, 3 * D * D},
    {ro_w1,  wb_ro1,  D * D},
  };
  for (int i = 0; i < 9; ++i)
    k_convert_bf16<<<cdiv(wc[i].n, 256), 256, 0, stream>>>(wc[i].s, wc[i].d, wc[i].n);

  zero(ldeg, Lr); zero(cdeg, C);
  k_degree<<<cdiv(E, 256), 256, 0, stream>>>(lEdge, ldeg, E);
  k_degree<<<cdiv(E, 256), 256, 0, stream>>>(cEdge, cdeg, E);

  const float invNorm = sqrtf(2.0f) / sqrtf((float)D);   // 1 / (sqrt(D)/sqrt(2))
  k_init_emb<<<cdiv((i64)Lr * D, 256), 256, 0, stream>>>(lInit, hl, hlb, Lr, D, invNorm);
  k_init_emb<<<cdiv((i64)C * D, 256), 256, 0, stream>>>(cInit, hc, hcb, C, D, invNorm);

  for (int it = 0; it < NITER; ++it) {
    // ---- literal -> clause ----
    gemm(hlb,  wb_l2c1, l2c_b1, nullptr, hbuf, Lr, D, D, 1);   // hidden = relu(...)
    gemm(hbuf, wb_l2c2, l2c_b2, mbuf, nullptr, Lr, D, D, 0);   // msg (f32)
    zero(abuf, (i64)C * D);
    k_scatter<<<4096, 256, 0, stream>>>(mbuf, lEdge, cEdge, abuf, E, D);
    k_div_deg_bf16<<<cdiv((i64)C * D, 256), 256, 0, stream>>>(abuf, cdeg, xbuf, C, D);
    gemm(xbuf, wb_cwih, cg_bih, g1, nullptr, C, D, 3 * D, 0);  // gi
    gemm(hcb,  wb_cwhh, cg_bhh, g2, nullptr, C, D, 3 * D, 0);  // gh
    k_gru<<<cdiv((i64)C * D, 256), 256, 0, stream>>>(g1, g2, hc, hcb, C, D);

    // ---- clause -> literal ----
    gemm(hcb,  wb_c2l1, c2l_b1, nullptr, hbuf, C, D, D, 1);
    gemm(hbuf, wb_c2l2, c2l_b2, mbuf, nullptr, C, D, D, 0);
    zero(abuf, (i64)Lr * D);
    k_scatter<<<4096, 256, 0, stream>>>(mbuf, cEdge, lEdge, abuf, E, D);
    k_build_lgru_x<<<cdiv((i64)Lr * D, 256), 256, 0, stream>>>(abuf, ldeg, hl, xbuf, Lr, D);
    gemm(xbuf, wb_lwih, lg_bih, g1, nullptr, Lr, 2 * D, 3 * D, 0);
    gemm(hlb,  wb_lwhh, lg_bhh, g2, nullptr, Lr, D, 3 * D, 0);
    k_gru<<<cdiv((i64)Lr * D, 256), 256, 0, stream>>>(g1, g2, hl, hlb, Lr, D);
  }

  // ---- readout ----
  gemm(hlb, wb_ro1, ro_b1, nullptr, hbuf, Lr, D, D, 1);
  zero(gsum, B); zero(gcnt, B);
  k_readout<<<cdiv(Lr, 256), 256, 0, stream>>>(hbuf, ro_w2, ro_b2, lBatch, gsum, gcnt, Lr, D);
  k_final<<<1, 64, 0, stream>>>(gsum, gcnt, (float*)d_out, B);
}
